// MotilityModel_27479200759931
// MI455X (gfx1250) — matrix operand order repaired and checked by reading, not compiled, not run on hardware
//
#include <hip/hip_runtime.h>

typedef __attribute__((ext_vector_type(16))) __bf16 v16bf;
typedef __attribute__((ext_vector_type(8)))  float  v8f;
typedef __attribute__((ext_vector_type(4)))  int    v4i;
typedef __attribute__((address_space(1))) v4i* gv4i_p;   // global int4*
typedef __attribute__((address_space(3))) v4i* lv4i_p;   // LDS int4*

#define N_CELLS 500000
#define CS_DIM  64
#define H_DIM   256
#define S_DIM   3
#define LN_EPS  1e-5f

#define ROWS_PER_WAVE   16
#define WAVES_PER_BLOCK 8
#define ROWS_PER_BLOCK  (ROWS_PER_WAVE * WAVES_PER_BLOCK)       // 128
#define LDS_STRIDE      72   // bf16 elems per W1 row: 64 payload + 8 pad (conflict-free b128 reads)
#define W1HALF_ELEMS    (H_DIM * LDS_STRIDE)                    // 18432 bf16 per hi/lo array
#define SMEM_BYTES      (2 * W1HALF_ELEMS * (int)sizeof(__bf16)) // 73728 B
#define COPY_CHUNKS     (SMEM_BYTES / 16)                       // 4608 b128 chunks = 18 * 256

#if defined(__has_builtin)
#if __has_builtin(__builtin_amdgcn_global_load_async_to_lds_b128)
#define HAVE_ASYNC_LDS 1
#endif
#endif

union B32x2 { uint4 q[2]; v16bf v; };

// ---- Prep: split W1 (f32) into bf16 hi/lo once, in the padded LDS layout ----
__global__ __launch_bounds__(256) void w1_split_kernel(const float* __restrict__ W1,
                                                       __bf16* __restrict__ ws) {
    int i = blockIdx.x * 256 + threadIdx.x;          // 0 .. 16383
    if (i < H_DIM * CS_DIM) {
        int j = i >> 6, c = i & 63;
        float f  = W1[i];
        __bf16 hi = (__bf16)f;
        __bf16 lo = (__bf16)(f - (float)hi);
        ws[j * LDS_STRIDE + c]                 = hi;
        ws[W1HALF_ELEMS + j * LDS_STRIDE + c]  = lo;
    }
}

__global__ __launch_bounds__(256) void motility_fused_kernel(
    const float*  __restrict__ cs,     // [N, 64]
    const __bf16* __restrict__ wsplit, // padded hi/lo W1, SMEM_BYTES bytes
    const float*  __restrict__ b1,     // [256]
    const float*  __restrict__ gam,    // [256]
    const float*  __restrict__ bet,    // [256]
    const float*  __restrict__ W2,     // [3, 256]
    const float*  __restrict__ b2,     // [3]
    float* __restrict__ out)           // [N, 3]
{
    extern __shared__ char smem[];
    __bf16* w1hi = (__bf16*)smem;
    __bf16* w1lo = w1hi + W1HALF_ELEMS;

    const int tid = threadIdx.x;

    // ---- Stage pre-split W1 into LDS: flat 73728-B copy, async global->LDS path ----
    #pragma unroll
    for (int it = 0; it < COPY_CHUNKS / 256; ++it) {     // 18 iterations, exact
        int idx = it * 256 + tid;
#ifdef HAVE_ASYNC_LDS
        __builtin_amdgcn_global_load_async_to_lds_b128(
            (gv4i_p)((const uint4*)wsplit + idx),
            (lv4i_p)((uint4*)smem + idx),
            0, 0);
#else
        ((uint4*)smem)[idx] = ((const uint4*)wsplit)[idx];
#endif
    }
#ifdef HAVE_ASYNC_LDS
#if __has_builtin(__builtin_amdgcn_s_wait_asynccnt)
    __builtin_amdgcn_s_wait_asynccnt(0);
#else
    asm volatile("s_wait_asynccnt 0" ::: "memory");
#endif
#endif
    __syncthreads();

    const int wave  = tid >> 5;
    const int lane  = tid & 31;
    const int lhalf = lane >> 4;        // 0: lanes 0-15, 1: lanes 16-31
    const int lmod  = lane & 15;

    const long rowbase = (long)blockIdx.x * ROWS_PER_BLOCK + (long)wave * ROWS_PER_WAVE;
    long rclamp = rowbase + lmod;
    if (rclamp > (long)(N_CELLS - 1)) rclamp = (long)(N_CELLS - 1);  // clamp reads; EXEC stays all-1s
    const float* rowp = cs + rclamp * CS_DIM;

    // ---- Load A tile (16 rows x 64 cols) and convert to bf16 hi/lo fragments ----
    // A layout (16-bit, 16x32): lane<16 holds K{0..7,16..23}, lane>=16 K{8..15,24..31}, pairs per VGPR.
    v16bf Ahi[2], Alo[2];
    #pragma unroll
    for (int s = 0; s < 2; ++s) {
        const int k0 = 32 * s + (lhalf ? 8  : 0);
        const int k1 = 32 * s + (lhalf ? 24 : 16);
        float4 q0 = *(const float4*)(rowp + k0);
        float4 q1 = *(const float4*)(rowp + k0 + 4);
        float4 q2 = *(const float4*)(rowp + k1);
        float4 q3 = *(const float4*)(rowp + k1 + 4);
        float f[16] = {q0.x,q0.y,q0.z,q0.w, q1.x,q1.y,q1.z,q1.w,
                       q2.x,q2.y,q2.z,q2.w, q3.x,q3.y,q3.z,q3.w};
        #pragma unroll
        for (int e = 0; e < 16; ++e) {
            __bf16 hi = (__bf16)f[e];
            Ahi[s][e] = hi;
            Alo[s][e] = (__bf16)(f[e] - (float)hi);
        }
    }

    // ---- Accumulators: 16 col-tiles of 16x16 f32, initialized with bias (col-dependent splat) ----
    v8f acc[16];
    #pragma unroll
    for (int t = 0; t < 16; ++t) {
        float bv = b1[t * 16 + lmod];
        #pragma unroll
        for (int e = 0; e < 8; ++e) acc[t][e] = bv;
    }

    // ---- GEMM1: 16 tiles x 2 K-steps x 3 split terms = 96 v_wmma_f32_16x16x32_bf16 ----
    #pragma unroll
    for (int t = 0; t < 16; ++t) {
        const int n = t * 16 + lmod;                      // B column for this lane
        #pragma unroll
        for (int s = 0; s < 2; ++s) {
            const int koff = 32 * s + (lhalf ? 8 : 0);    // B: lane<16 K{0..7,16..23}, lane>=16 K{8..15,24..31}
            B32x2 bh, bl;
            const uint4* ph = (const uint4*)(w1hi + n * LDS_STRIDE + koff);
            const uint4* pl = (const uint4*)(w1lo + n * LDS_STRIDE + koff);
            bh.q[0] = ph[0]; bh.q[1] = ph[2];
            bl.q[0] = pl[0]; bl.q[1] = pl[2];
            acc[t] = __builtin_amdgcn_wmma_f32_16x16x32_bf16(
                         false, Ahi[s], false, bh.v, (short)0, acc[t], false, false);
            acc[t] = __builtin_amdgcn_wmma_f32_16x16x32_bf16(
                         false, Alo[s], false, bh.v, (short)0, acc[t], false, false);
            acc[t] = __builtin_amdgcn_wmma_f32_16x16x32_bf16(
                         false, Ahi[s], false, bl.v, (short)0, acc[t], false, false);
        }
    }

    // ---- LayerNorm stats: per-lane partials, then reduce across the 16-lane column group ----
    float rs[8], rq[8];
    #pragma unroll
    for (int v = 0; v < 8; ++v) { rs[v] = 0.f; rq[v] = 0.f; }
    #pragma unroll
    for (int t = 0; t < 16; ++t)
        #pragma unroll
        for (int v = 0; v < 8; ++v) {
            float x = acc[t][v];
            rs[v] += x;
            rq[v]  = fmaf(x, x, rq[v]);
        }
    #pragma unroll
    for (int off = 1; off < 16; off <<= 1) {
        #pragma unroll
        for (int v = 0; v < 8; ++v) {
            rs[v] += __shfl_xor(rs[v], off, 32);   // xor masks <16 stay within row-group
            rq[v] += __shfl_xor(rq[v], off, 32);
        }
    }
    float rstd[8], nms[8];                          // rstd, and -mean*rstd
    #pragma unroll
    for (int v = 0; v < 8; ++v) {
        float mean = rs[v] * (1.0f / H_DIM);
        float var  = fmaf(-mean, mean, rq[v] * (1.0f / H_DIM));
        rstd[v] = rsqrtf(var + LN_EPS);
        nms[v]  = -mean * rstd[v];
    }

    // ---- Normalize + affine + ReLU fused with GEMM2 (256 -> 3): 6 VALU ops per element ----
    float fo[3][8];
    #pragma unroll
    for (int s = 0; s < 3; ++s)
        #pragma unroll
        for (int v = 0; v < 8; ++v) fo[s][v] = 0.f;

    #pragma unroll
    for (int t = 0; t < 16; ++t) {
        const int n = t * 16 + lmod;
        float g   = gam[n], be = bet[n];
        float w0  = W2[0 * H_DIM + n];
        float w1v = W2[1 * H_DIM + n];
        float w2v = W2[2 * H_DIM + n];
        #pragma unroll
        for (int v = 0; v < 8; ++v) {
            float h = fmaf(fmaf(acc[t][v], rstd[v], nms[v]), g, be);
            h = fmaxf(h, 0.0f);
            fo[0][v] = fmaf(h, w0,  fo[0][v]);
            fo[1][v] = fmaf(h, w1v, fo[1][v]);
            fo[2][v] = fmaf(h, w2v, fo[2][v]);
        }
    }
    #pragma unroll
    for (int off = 1; off < 16; off <<= 1)
        #pragma unroll
        for (int s = 0; s < 3; ++s)
            #pragma unroll
            for (int v = 0; v < 8; ++v)
                fo[s][v] += __shfl_xor(fo[s][v], off, 32);

    // ---- Store: lane 0 writes rows base+0..7, lane 16 writes rows base+8..15 ----
    if (lmod == 0) {
        float bb0 = b2[0], bb1 = b2[1], bb2 = b2[2];
        #pragma unroll
        for (int v = 0; v < 8; ++v) {
            long rr = rowbase + v + 8 * lhalf;
            if (rr < (long)N_CELLS) {
                out[rr * 3 + 0] = fo[0][v] + bb0;
                out[rr * 3 + 1] = fo[1][v] + bb1;
                out[rr * 3 + 2] = fo[2][v] + bb2;
            }
        }
    }
}

extern "C" void kernel_launch(void* const* d_in, const int* in_sizes, int n_in,
                              void* d_out, int out_size, void* d_ws, size_t ws_size,
                              hipStream_t stream) {
    (void)in_sizes; (void)n_in; (void)ws_size; (void)out_size;
    const float* cs  = (const float*)d_in[0];
    const float* W1  = (const float*)d_in[1];
    const float* b1  = (const float*)d_in[2];
    const float* gam = (const float*)d_in[3];
    const float* bet = (const float*)d_in[4];
    const float* W2  = (const float*)d_in[5];
    const float* b2  = (const float*)d_in[6];
    float* out       = (float*)d_out;
    __bf16* wsplit   = (__bf16*)d_ws;                 // 73728 B of scratch used

    // One-time (per launch) W1 hi/lo split into scratch, padded to the LDS layout.
    w1_split_kernel<<<dim3((H_DIM * CS_DIM + 255) / 256), dim3(256), 0, stream>>>(W1, wsplit);

    const int grid = (N_CELLS + ROWS_PER_BLOCK - 1) / ROWS_PER_BLOCK;  // 3907
    motility_fused_kernel<<<dim3(grid), dim3(256), SMEM_BYTES, stream>>>(
        cs, wsplit, b1, gam, bet, W2, b2, out);
}
